// CarefulSparseResNet3D_72962904425005
// MI455X (gfx1250) — matrix operand order repaired and checked by reading, not compiled, not run on hardware
//
#include <hip/hip_runtime.h>

// ---------------------------------------------------------------------------
// MI455X (gfx1250): every conv as implicit GEMM on v_wmma_f32_16x16x32_bf16
// (wave32, f32 accumulate). fp32 residual trunk (NCDHW), bf16 intermediates
// in channels-innermost [b][spatial][ci] layout, stored pre-masked so the
// consumers can stage them into LDS with GLOBAL_LOAD_ASYNC_TO_LDS_B128
// (ASYNCcnt path, per-lane LDS destinations keep the padded, conflict-free
// LDS stride). Weights pre-packed into exact A-fragment order.
// Register blocking: one A-fragment load feeds two WMMAs.
// ---------------------------------------------------------------------------

typedef __bf16 bf16_t;
typedef __bf16 v16bf __attribute__((ext_vector_type(16)));
typedef __bf16 v8bf  __attribute__((ext_vector_type(8)));
typedef float  v8f   __attribute__((ext_vector_type(8)));

#define WMMA_BF16(acc, a, b)                                                   \
  acc = __builtin_amdgcn_wmma_f32_16x16x32_bf16(false, (a), false, (b),        \
                                                (short)0, (acc), false, false)

// Async 16B copy global -> LDS (per-lane LDS destination), ASYNCcnt-tracked.
#define ASYNC_CP_B128(ldsoff, gbase, voff)                                     \
  asm volatile("global_load_async_to_lds_b128 %0, %1, %2"                      \
               ::"v"(ldsoff), "v"(voff), "s"(gbase)                            \
               : "memory")

__device__ __forceinline__ void wait_async0() {
#if __has_builtin(__builtin_amdgcn_s_wait_asynccnt)
  __builtin_amdgcn_s_wait_asynccnt(0);
#else
  asm volatile("s_wait_asynccnt 0x0" ::: "memory");
#endif
}

// A-fragment (16x32 bf16) k-index for (lane, element j): contiguous k = 16*(lane/16) + j
__device__ __forceinline__ int kA_of(int lane, int j) {
  return ((lane >> 4) << 4) + j;
}

__device__ __forceinline__ float bn_eval(float x, int c, const float* g,
                                         const float* b, const float* m,
                                         const float* v) {
  float inv = rsqrtf(v[c] + 1e-5f);
  return (x - m[c]) * inv * g[c] + b[c];
}
__device__ __forceinline__ float lrelu_f(float x) {
  return x > 0.f ? x : 0.01f * x;
}

// ---------------------------------------------------------------------------
// Weight repack: W[64][Cin][T] fp32 -> bf16 A fragments:
// packed[q][co_tile][lane][j] = W[co_tile*16 + lane%16][k % Cin][k / Cin],
// k = q*32 + kA(lane,j).  One 32B load per lane per A fragment.
// ---------------------------------------------------------------------------
__global__ void pack_w_kernel(const float* __restrict__ W,
                              bf16_t* __restrict__ out, int Cin, int T,
                              int total) {
  int idx = blockIdx.x * blockDim.x + threadIdx.x;
  if (idx >= total) return;
  int j = idx & 15;
  int lane = (idx >> 4) & 31;
  int ct = (idx >> 9) & 3;
  int q = idx >> 11;
  int kk = q * 32 + kA_of(lane, j);
  int ci = kk % Cin;
  int tap = kk / Cin;
  int co = ct * 16 + (lane & 15);
  out[idx] = (bf16_t)W[(co * Cin + ci) * T + tap];
}

// ---------------------------------------------------------------------------
// Mask downsample: maxpool 2x2x2 + binarize. Writes ws copy and output copy.
// ---------------------------------------------------------------------------
__global__ void maskdown_kernel(const float* __restrict__ mask,
                                float* __restrict__ dmask,
                                float* __restrict__ outMask) {
  int idx = blockIdx.x * blockDim.x + threadIdx.x;
  if (idx >= 4 * 32768) return;
  int b = idx >> 15;
  int r = idx & 32767;
  int z = r >> 10, y = (r >> 5) & 31, x = r & 31;
  float mx = 0.f;
  for (int dz = 0; dz < 2; ++dz)
    for (int dy = 0; dy < 2; ++dy)
      for (int dx = 0; dx < 2; ++dx) {
        float v = mask[(size_t)b * 262144 + (2 * z + dz) * 4096 +
                       (2 * y + dy) * 64 + (2 * x + dx)];
        mx = fmaxf(mx, v);
      }
  float v = (mx > 0.5f) ? 1.f : 0.f;
  dmask[idx] = v;
  outMask[idx] = v;
}

// mc3 = 27 / max(conv3d(dmask, ones3, pad 1), 1e-5)
__global__ void mc3_kernel(const float* __restrict__ dmask,
                           float* __restrict__ mc3) {
  int idx = blockIdx.x * blockDim.x + threadIdx.x;
  if (idx >= 4 * 32768) return;
  int b = idx >> 15;
  int r = idx & 32767;
  int z = r >> 10, y = (r >> 5) & 31, x = r & 31;
  float s = 0.f;
  for (int dz = -1; dz <= 1; ++dz)
    for (int dy = -1; dy <= 1; ++dy)
      for (int dx = -1; dx <= 1; ++dx) {
        int zz = z + dz, yy = y + dy, xx = x + dx;
        if (zz >= 0 && zz < 32 && yy >= 0 && yy < 32 && xx >= 0 && xx < 32)
          s += dmask[(b << 15) + (zz << 10) + (yy << 5) + xx];
      }
  mc3[idx] = 27.f / fmaxf(s, 1e-5f);
}

// ---------------------------------------------------------------------------
// Downsample conv: 4x4x4, stride 2, pad 1, Cin=32 -> Cout=64 (fp32 in, NCDHW).
// One WG computes two (b,z,y) output rows; each wave owns one co-tile + one
// x-half and both y rows: one A load feeds two WMMAs.
// Epilogue: +bias, *mc (64/sum window mask), BN, leaky relu -> fp32 NCDHW.
// ---------------------------------------------------------------------------
__global__ void __launch_bounds__(256) down_kernel(
    const float* __restrict__ feat, const float* __restrict__ mask,
    const bf16_t* __restrict__ pW, const float* __restrict__ bias,
    const float* __restrict__ g, const float* __restrict__ bb,
    const float* __restrict__ bm, const float* __restrict__ bv,
    float* __restrict__ out) {
  __shared__ bf16_t sIn[6 * 66 * 48];  // [dyy 0..5][xx 0..65][ci], stride 48
  __shared__ float sMask[4 * 6 * 66];  // [dz][dyy][xx]
  __shared__ float sMC[64];            // [yo][X]

  int bx = blockIdx.x;
  int b = bx >> 9, z = (bx >> 4) & 31, y0 = (bx & 15) * 2;
  int tid = threadIdx.x;
  int wid = tid >> 5, lane = tid & 31;
  int co_t = wid & 3, nh = wid >> 2;
  int X = nh * 16 + (lane & 15);
  int cib = (lane >> 4) << 4;

  for (int rr = tid; rr < 24; rr += 256) {
    int dz = rr / 6, dyy = rr % 6;
    int iz = 2 * z + dz - 1, iy = 2 * y0 + dyy - 1;
    bool ok = (iz >= 0 && iz < 64 && iy >= 0 && iy < 64);
    const float* mrow = mask + (size_t)b * 262144 + iz * 4096 + iy * 64;
    for (int xx = 0; xx < 66; ++xx) {
      int ix = xx - 1;
      sMask[rr * 66 + xx] = (ok && ix >= 0 && ix < 64) ? mrow[ix] : 0.f;
    }
  }
  __syncthreads();
  if (tid < 64) {
    int yo = tid >> 5, Xs = tid & 31;
    float s = 0.f;
    for (int dz = 0; dz < 4; ++dz)
      for (int dy = 0; dy < 4; ++dy)
        for (int dx = 0; dx < 4; ++dx)
          s += sMask[(dz * 6 + dy + 2 * yo) * 66 + 2 * Xs + dx];
    sMC[tid] = 64.f / fmaxf(s, 1e-5f);
  }

  v8f acc0 = {0.f, 0.f, 0.f, 0.f, 0.f, 0.f, 0.f, 0.f};
  v8f acc1 = acc0;

  for (int dz = 0; dz < 4; ++dz) {
    __syncthreads();
    for (int rr = tid; rr < 192; rr += 256) {
      int dyy = rr >> 5, ci = rr & 31;
      int iz = 2 * z + dz - 1, iy = 2 * y0 + dyy - 1;
      bool ok = (iz >= 0 && iz < 64 && iy >= 0 && iy < 64);
      const float* frow =
          feat + ((size_t)(b * 32 + ci) * 64 + iz) * 4096 + iy * 64;
      for (int xx = 0; xx < 66; ++xx) {
        int ix = xx - 1;
        float vv = (ok && ix >= 0 && ix < 64) ? frow[ix] : 0.f;
        sIn[(dyy * 66 + xx) * 48 + ci] = (bf16_t)vv;
      }
    }
    __syncthreads();
    for (int dy = 0; dy < 4; ++dy) {
#pragma unroll
      for (int kc = 0; kc < 4; ++kc) {  // kc == dx
        int aoff = ((((dz * 4 + dy) * 4 + kc) * 4 + co_t) * 32 + lane) * 16;
        v16bf a = *(const v16bf*)(pW + aoff);
        int xx = 2 * X + kc;  // ix = 2X+dx-1, xx = ix+1
        v16bf b0 = *(const v16bf*)(&sIn[((dy + 0) * 66 + xx) * 48 + cib]);
        WMMA_BF16(acc0, a, b0);
        v16bf b1 = *(const v16bf*)(&sIn[((dy + 2) * 66 + xx) * 48 + cib]);
        WMMA_BF16(acc1, a, b1);
      }
    }
  }

#pragma unroll
  for (int yo = 0; yo < 2; ++yo) {
    int yy = y0 + yo;
    float mc = sMC[yo * 32 + X];
#pragma unroll
    for (int r = 0; r < 8; ++r) {
      int co = co_t * 16 + ((lane >> 4) << 3) + r;
      float val = (yo ? acc1[r] : acc0[r]) + bias[co];
      val *= mc;
      val = bn_eval(val, co, g, bb, bm, bv);
      val = lrelu_f(val);
      out[((size_t)(b * 64 + co) << 15) + (z << 10) + (yy << 5) + X] = val;
    }
  }
}

// ---------------------------------------------------------------------------
// 3x3x3 submanifold conv, C=64->64, pad 1.
// Input Xin: bf16 [b][spatial][ci], PRE-MASKED -> staged with async copies
// (per-lane LDS dest keeps padded stride; halos zero-filled by DS stores).
// Output Y: bf16 [b][spatial][ci], pre-masked for the c3 GEMM (one 16B
// vector store per lane per y-row).
// Epilogue: +bias, *mask, *mc3, BN, leaky relu, *mask(store).
// ---------------------------------------------------------------------------
__global__ void __launch_bounds__(256) conv3_kernel(
    const bf16_t* __restrict__ Xin, const float* __restrict__ dmask,
    const float* __restrict__ mc3, const bf16_t* __restrict__ pW,
    const float* __restrict__ bias, const float* __restrict__ g,
    const float* __restrict__ bb, const float* __restrict__ bm,
    const float* __restrict__ bv, bf16_t* __restrict__ Y) {
  __shared__ bf16_t sIn[12 * 34 * 72];  // [dz 0..2][dyy 0..3][xx][ci], str 72

  int bx = blockIdx.x;
  int b = bx >> 9, z = (bx >> 4) & 31, y0 = (bx & 15) * 2;
  int tid = threadIdx.x;
  unsigned long long gbase = (unsigned long long)Xin;

  // Zero the halo columns xx = 0 and 33 (async copies only write xx 1..32).
  for (int d = tid; d < 768; d += 256) {
    int czy = d >> 6, rem = d & 63;
    int xx = (rem >> 5) ? 33 : 0;
    *(unsigned*)&sIn[(czy * 34 + xx) * 72 + (rem & 31) * 2] = 0u;
  }
  // Async-stage 12 (iz,iy) rows of [32 x][64 ci]; zero-fill OOB rows.
  for (int czy = 0; czy < 12; ++czy) {
    int dz = czy >> 2, dyy = czy & 3;
    int iz = z + dz - 1, iy = y0 + dyy - 1;
    if (iz >= 0 && iz < 32 && iy >= 0 && iy < 32) {
      int x = tid >> 3, seg = tid & 7;  // 256 chunks of 16B
      unsigned voff =
          (unsigned)((((b << 15) + (iz << 10) + (iy << 5) + x) << 7) +
                     (seg << 4));
      unsigned ldsoff =
          (unsigned)(size_t)(&sIn[(czy * 34 + x + 1) * 72 + seg * 8]);
      ASYNC_CP_B128(ldsoff, gbase, voff);
    } else {
      for (int d = tid; d < 1088; d += 256) {
        int xx = d >> 5;
        *(unsigned*)&sIn[(czy * 34 + xx) * 72 + (d & 31) * 2] = 0u;
      }
    }
  }
  wait_async0();
  __syncthreads();

  int wid = tid >> 5, lane = tid & 31;
  int co_t = wid & 3, nh = wid >> 2;
  int X = nh * 16 + (lane & 15);
  v8f acc0 = {0.f, 0.f, 0.f, 0.f, 0.f, 0.f, 0.f, 0.f};
  v8f acc1 = acc0;

  for (int dz = 0; dz < 3; ++dz) {
    for (int dy = 0; dy < 3; ++dy) {
#pragma unroll
      for (int dx = 0; dx < 3; ++dx) {
        int xx = X + dx;  // ix = X+dx-1, xx = ix+1
#pragma unroll
        for (int kc = 0; kc < 2; ++kc) {
          int tap = (dz * 3 + dy) * 3 + dx;
          int aoff = (((tap * 2 + kc) * 4 + co_t) * 32 + lane) * 16;
          v16bf a = *(const v16bf*)(pW + aoff);
          int cib = kc * 32 + ((lane >> 4) << 4);
          v16bf b0 =
              *(const v16bf*)(&sIn[((dz * 4 + dy + 0) * 34 + xx) * 72 + cib]);
          WMMA_BF16(acc0, a, b0);
          v16bf b1 =
              *(const v16bf*)(&sIn[((dz * 4 + dy + 1) * 34 + xx) * 72 + cib]);
          WMMA_BF16(acc1, a, b1);
        }
      }
    }
  }

#pragma unroll
  for (int yo = 0; yo < 2; ++yo) {
    int yy = y0 + yo;
    int sp = ((int)b << 15) + (z << 10) + (yy << 5) + X;
    float dm = dmask[sp];
    float mk = dm * mc3[sp];
    v8bf ov;
#pragma unroll
    for (int r = 0; r < 8; ++r) {
      int co = co_t * 16 + ((lane >> 4) << 3) + r;
      float val = ((yo ? acc1[r] : acc0[r]) + bias[co]) * mk;
      val = bn_eval(val, co, g, bb, bm, bv);
      val = lrelu_f(val);
      ov[r] = (bf16_t)(val * dm);  // pre-mask for the c3 GEMM
    }
    *(v8bf*)(&Y[(size_t)sp * 64 + co_t * 16 + ((lane >> 4) << 3)]) = ov;
  }
}

// ---------------------------------------------------------------------------
// 1x1x1 conv as GEMM: M=64, K=64, N=256 flat spatial per WG.
// Each wave pins a co-tile, preloads both A fragments once, sweeps 8 N-tiles.
// MODE 0: y = lrelu(bn((Wx+b)*mask)) * mask   (c1; fp32 NCDHW in, bf16
//         [sp][ci] pre-masked out; manual masked staging)
// MODE 1: y = lrelu(res + bn((Wx+b)*mask))    (c3; bf16 [sp][ci] pre-masked
//         in -> ASYNC staged; fp32 NCDHW out, in-place residual)
// MODE 2: y = bn(Wx+b) * mask                 (final; fp32 NCDHW in/out)
// ---------------------------------------------------------------------------
template <int MODE, typename TIN, typename TOUT>
__global__ void __launch_bounds__(256) gemm1x1_kernel(
    const TIN* __restrict__ Xin, const float* __restrict__ dmask,
    const bf16_t* __restrict__ pW, const float* __restrict__ bias,
    const float* __restrict__ g, const float* __restrict__ bb,
    const float* __restrict__ bm, const float* __restrict__ bv,
    const float* __restrict__ res, TOUT* __restrict__ Y) {
  __shared__ bf16_t sIn[256 * 80];  // [n][ci], ci stride 80

  int b = blockIdx.x >> 7;
  int nbase = (blockIdx.x & 127) << 8;
  int tid = threadIdx.x;
  int wid = tid >> 5, lane = tid & 31;
  int co_t = wid & 3, start = wid >> 2;

  // A fragments: load once per wave (independent of LDS staging)
  v16bf a0 = *(const v16bf*)(pW + ((0 * 4 + co_t) * 32 + lane) * 16);
  v16bf a1 = *(const v16bf*)(pW + ((1 * 4 + co_t) * 32 + lane) * 16);

  if (MODE == 1) {
    // Pre-masked bf16 [sp][ci] input: straight async copy into padded LDS.
    unsigned long long gb = (unsigned long long)Xin;
    for (int c = tid; c < 2048; c += 256) {
      int n = c >> 3, seg = c & 7;
      unsigned voff = (unsigned)((((b << 15) + nbase + n) << 7) + (seg << 4));
      unsigned ldsoff = (unsigned)(size_t)(&sIn[n * 80 + seg * 8]);
      ASYNC_CP_B128(ldsoff, gb, voff);
    }
    wait_async0();
  } else {
    for (int idx = tid; idx < 64 * 256; idx += 256) {
      int ci = idx >> 8, nn = idx & 255;
      int sp = (b << 15) + nbase + nn;
      float vv =
          (float)Xin[((size_t)(b * 64 + ci) << 15) + nbase + nn] * dmask[sp];
      sIn[nn * 80 + ci] = (bf16_t)vv;
    }
  }
  __syncthreads();

  int cib = (lane >> 4) << 4;
  for (int ntile = start; ntile < 16; ntile += 2) {
    int np = ntile * 16 + (lane & 15);
    v8f acc = {0.f, 0.f, 0.f, 0.f, 0.f, 0.f, 0.f, 0.f};
    v16bf b0 = *(const v16bf*)(&sIn[np * 80 + cib]);
    WMMA_BF16(acc, a0, b0);
    v16bf b1 = *(const v16bf*)(&sIn[np * 80 + 32 + cib]);
    WMMA_BF16(acc, a1, b1);

    int sp = (b << 15) + nbase + np;
    float mk = dmask[sp];
    if (MODE == 0) {
      v8bf ov;
#pragma unroll
      for (int r = 0; r < 8; ++r) {
        int co = co_t * 16 + ((lane >> 4) << 3) + r;
        float val = (acc[r] + bias[co]) * mk;
        val = bn_eval(val, co, g, bb, bm, bv);
        val = lrelu_f(val);
        ov[r] = (bf16_t)(val * mk);  // pre-mask for conv3
      }
      bf16_t* yb = (bf16_t*)Y;
      *(v8bf*)(&yb[(size_t)sp * 64 + co_t * 16 + ((lane >> 4) << 3)]) = ov;
    } else if (MODE == 1) {
#pragma unroll
      for (int r = 0; r < 8; ++r) {
        int co = co_t * 16 + ((lane >> 4) << 3) + r;
        float val = (acc[r] + bias[co]) * mk;
        val = bn_eval(val, co, g, bb, bm, bv);
        val += res[((size_t)(b * 64 + co) << 15) + nbase + np];
        val = lrelu_f(val);
        Y[((size_t)(b * 64 + co) << 15) + nbase + np] = (TOUT)val;
      }
    } else {
#pragma unroll
      for (int r = 0; r < 8; ++r) {
        int co = co_t * 16 + ((lane >> 4) << 3) + r;
        float val = bn_eval(acc[r] + bias[co], co, g, bb, bm, bv);
        val *= mk;
        Y[((size_t)(b * 64 + co) << 15) + nbase + np] = (TOUT)val;
      }
    }
  }
}

// ---------------------------------------------------------------------------
extern "C" void kernel_launch(void* const* d_in, const int* in_sizes, int n_in,
                              void* d_out, int out_size, void* d_ws,
                              size_t ws_size, hipStream_t stream) {
  (void)in_sizes; (void)n_in; (void)out_size; (void)ws_size;
  const float* feat = (const float*)d_in[0];
  const float* mask = (const float*)d_in[1];
  const float* down_w = (const float*)d_in[2];
  const float* down_b = (const float*)d_in[3];
  const float* dbn[4] = {(const float*)d_in[4], (const float*)d_in[5],
                         (const float*)d_in[6], (const float*)d_in[7]};
  const float* fin_w = (const float*)d_in[62];
  const float* fin_b = (const float*)d_in[63];
  const float* fbn[4] = {(const float*)d_in[64], (const float*)d_in[65],
                         (const float*)d_in[66], (const float*)d_in[67]};

  char* ws = (char*)d_ws;
  size_t off = 0;
  auto alloc = [&](size_t bytes) {
    void* p = ws + off;
    off += (bytes + 255) & ~(size_t)255;
    return p;
  };
  const size_t SP = 32768;  // 32^3
  float* featA = (float*)alloc(4 * 64 * SP * 4);
  bf16_t* t1 = (bf16_t*)alloc(4 * 64 * SP * 2);   // [b][sp][ci] pre-masked
  bf16_t* t2 = (bf16_t*)alloc(4 * 64 * SP * 2);   // [b][sp][ci] pre-masked
  float* dmask = (float*)alloc(4 * SP * 4);
  float* mc3 = (float*)alloc(4 * SP * 4);
  bf16_t* pW_down = (bf16_t*)alloc(64 * 2048 * 2);
  bf16_t* pW_c2[3];
  for (int i = 0; i < 3; ++i) pW_c2[i] = (bf16_t*)alloc(64 * 1728 * 2);
  bf16_t* pW_11[7];  // 0..2: c1 blocks, 3..5: c3 blocks, 6: fin
  for (int i = 0; i < 7; ++i) pW_11[i] = (bf16_t*)alloc(64 * 64 * 2);

  // ---- weight repack ----
  {
    int total = 64 * 32 * 64;
    pack_w_kernel<<<(total + 255) / 256, 256, 0, stream>>>(down_w, pW_down,
                                                           32, 64, total);
  }
  for (int i = 0; i < 3; ++i) {
    int base = 8 + 18 * i;
    int total = 64 * 64 * 27;
    pack_w_kernel<<<(total + 255) / 256, 256, 0, stream>>>(
        (const float*)d_in[base + 2], pW_c2[i], 64, 27, total);
    int t11 = 64 * 64;
    pack_w_kernel<<<(t11 + 255) / 256, 256, 0, stream>>>(
        (const float*)d_in[base + 0], pW_11[i], 64, 1, t11);
    pack_w_kernel<<<(t11 + 255) / 256, 256, 0, stream>>>(
        (const float*)d_in[base + 4], pW_11[3 + i], 64, 1, t11);
  }
  {
    int t11 = 64 * 64;
    pack_w_kernel<<<(t11 + 255) / 256, 256, 0, stream>>>(fin_w, pW_11[6], 64,
                                                         1, t11);
  }

  // ---- masks ----
  float* outF = (float*)d_out;
  float* outM = outF + 4 * 64 * SP;
  maskdown_kernel<<<512, 256, 0, stream>>>(mask, dmask, outM);
  mc3_kernel<<<512, 256, 0, stream>>>(dmask, mc3);

  // ---- downsample conv ----
  down_kernel<<<2048, 256, 0, stream>>>(feat, mask, pW_down, down_b, dbn[0],
                                        dbn[1], dbn[2], dbn[3], featA);

  // ---- residual blocks ----
  for (int i = 0; i < 3; ++i) {
    int base = 8 + 18 * i;
    const float* c1_b = (const float*)d_in[base + 1];
    const float* c2_b = (const float*)d_in[base + 3];
    const float* c3_b = (const float*)d_in[base + 5];
    const float* bn1[4] = {(const float*)d_in[base + 6],
                           (const float*)d_in[base + 7],
                           (const float*)d_in[base + 8],
                           (const float*)d_in[base + 9]};
    const float* bn2[4] = {(const float*)d_in[base + 10],
                           (const float*)d_in[base + 11],
                           (const float*)d_in[base + 12],
                           (const float*)d_in[base + 13]};
    const float* bn3[4] = {(const float*)d_in[base + 14],
                           (const float*)d_in[base + 15],
                           (const float*)d_in[base + 16],
                           (const float*)d_in[base + 17]};
    gemm1x1_kernel<0, float, bf16_t><<<512, 256, 0, stream>>>(
        featA, dmask, pW_11[i], c1_b, bn1[0], bn1[1], bn1[2], bn1[3], nullptr,
        t1);
    conv3_kernel<<<2048, 256, 0, stream>>>(t1, dmask, mc3, pW_c2[i], c2_b,
                                           bn2[0], bn2[1], bn2[2], bn2[3], t2);
    gemm1x1_kernel<1, bf16_t, float><<<512, 256, 0, stream>>>(
        t2, dmask, pW_11[3 + i], c3_b, bn3[0], bn3[1], bn3[2], bn3[3], featA,
        featA);
  }

  // ---- final 1x1 + BN + mask ----
  gemm1x1_kernel<2, float, float><<<512, 256, 0, stream>>>(
      featA, dmask, pW_11[6], fin_b, fbn[0], fbn[1], fbn[2], fbn[3], nullptr,
      outF);
}
